// MultiPrototypeContrastiveLoss_52493090291884
// MI455X (gfx1250) — matrix operand order repaired and checked by reading, not compiled, not run on hardware
//
#include <hip/hip_runtime.h>

// MultiPrototypeContrastiveLoss for MI455X (gfx1250, wave32, WMMA).
// Q = v_proj flattened (32768 x 128), P = prototypes flattened (2048 x 128).
// Fused bf16 WMMA GEMM + per-concept softmax + LSE + masked mean.
//
// GEMM orientation: A = prototype tile (16 rows from LDS), B = query tiles
// (registers). C layout gives each lane ALL 8 prototypes of one concept
// (acc[0..7]) for one query -> softmax over M is per-lane, no shuffles.
// Each wave carries TWO query tiles sharing the same A fragments so the
// ds_load_b128 : v_wmma ratio is 1:1 (LDS no longer the co-limiter).
// Proto tiles stream to LDS via global_load_async_to_lds_b128 (ASYNCcnt),
// double buffered, from prototypes pre-converted to bf16 once per launch.

namespace {
constexpr int Bq = 128, Kc = 256, Mp = 8, Dd = 128;
constexpr int NQ = Bq * Kc;     // 32768 queries
constexpr int NP = Kc * Mp;     // 2048 prototype rows
constexpr int NT = NP / 16;     // 128 prototype tiles (2 concepts each)
constexpr float kLambda = 20.0f, kGamma = 20.0f, kMargin = 0.05f, kEps = 1e-8f;
constexpr float kGL2E = 28.853900817779268f;   // 20 * log2(e), gamma==lambda==20
constexpr int LDS_ROW = 136;    // bf16 elems per LDS row (272 B stride)

typedef __attribute__((ext_vector_type(16))) __bf16 v16bf;
typedef __attribute__((ext_vector_type(16))) unsigned short v16u;
typedef __attribute__((ext_vector_type(8)))  unsigned short v8u;
typedef __attribute__((ext_vector_type(8)))  float v8f;
typedef __attribute__((ext_vector_type(4)))  float f4v;

__device__ __forceinline__ unsigned short f2bf(float f) {
  unsigned u = __builtin_bit_cast(unsigned, f);
  u += 0x7fffu + ((u >> 16) & 1u);        // RNE fp32 -> bf16
  return (unsigned short)(u >> 16);
}

constexpr unsigned dpp8sel(int x) {
  unsigned s = 0;
  for (int i = 0; i < 8; ++i) s |= (unsigned)((i ^ x) & 7) << (3 * i);
  return s;
}

template <int X>
__device__ __forceinline__ float dppx(float v) {
  return __builtin_bit_cast(
      float, __builtin_amdgcn_mov_dpp8(__builtin_bit_cast(int, v), dpp8sel(X)));
}

template <int X>  // ds_swizzle group-of-32 xor: imm = (X<<10)|0x1F
__device__ __forceinline__ float swzx(float v) {
  return __builtin_bit_cast(
      float,
      __builtin_amdgcn_ds_swizzle(__builtin_bit_cast(int, v), (X << 10) | 0x1F));
}
} // namespace

// Prototypes pre-converted to bf16, row-major (NP x Dd). Rebuilt every launch.
__device__ unsigned short g_pbf[NP * Dd];

__global__ __launch_bounds__(256) void mpcl_convert(const float* __restrict__ protos) {
  const int base = (blockIdx.x * 256 + threadIdx.x) * 8;   // 128 blocks cover NP*Dd
  const f4v* gp = (const f4v*)(protos + base);
  f4v a = gp[0], b = gp[1];
  float t[8] = {a.x, a.y, a.z, a.w, b.x, b.y, b.z, b.w};
  v8u pk;
#pragma unroll
  for (int e = 0; e < 8; ++e) pk[e] = f2bf(t[e]);
  *(v8u*)&g_pbf[base] = pk;
}

__global__ __launch_bounds__(256) void mpcl_main(
    const float* __restrict__ vproj, const int* __restrict__ labels,
    float* __restrict__ ws) {
  __shared__ __align__(16) unsigned short ldsP[2][16 * LDS_ROW];

  const int tid  = threadIdx.x;
  const int wave = tid >> 5;
  const int lane = tid & 31;
  const int h    = lane >> 4;   // lane half: selects concept within tile
  const int ln   = lane & 15;   // query column / proto row within tile
  const int qt0  = blockIdx.x * 16 + wave * 2;   // two query tiles per wave
  const int q0   = qt0 * 16 + ln;                // lane's query in tile 0
  const int q1   = q0 + 16;                      // lane's query in tile 1

  // ---- Preload B fragments once: 2 x (16 queries x 128 K), bf16.
  // B 32x16 layout per step s: lane holds column ln, K = 32s+8h+[0..8) then 32s+16+8h+[0..8).
  v16u qfr0[4], qfr1[4];
#pragma unroll
  for (int t = 0; t < 2; ++t) {
    const float* qrow = vproj + (size_t)(t ? q1 : q0) * Dd;
#pragma unroll
    for (int s = 0; s < 4; ++s) {
      const f4v* p = (const f4v*)(qrow + 32 * s + 8 * h);
      f4v x0 = p[0], x1 = p[1], x2 = p[4], x3 = p[5];
      float tt[16] = {x0.x, x0.y, x0.z, x0.w, x1.x, x1.y, x1.z, x1.w,
                      x2.x, x2.y, x2.z, x2.w, x3.x, x3.y, x3.z, x3.w};
      v16u* dst = t ? &qfr1[s] : &qfr0[s];
#pragma unroll
      for (int e = 0; e < 16; ++e) (*dst)[e] = f2bf(tt[e]);
    }
  }

  // ---- Async proto-tile copy: each thread moves one 16B chunk per tile.
  const int crow   = tid >> 4;          // 0..15 tile row
  const int cchunk = tid & 15;          // 0..15 16B chunk within row
  const unsigned ldsbase0 = (unsigned)(size_t)&ldsP[0][0];
  const unsigned ldsbase1 = (unsigned)(size_t)&ldsP[1][0];
  const unsigned dstoff   = (unsigned)(crow * 272 + cchunk * 16);
  const unsigned long long sbase = (unsigned long long)(size_t)&g_pbf[0];

#define ISSUE_COPY(ct, buf)                                                    \
  do {                                                                         \
    unsigned dst_  = ((buf) ? ldsbase1 : ldsbase0) + dstoff;                   \
    unsigned voff_ = (unsigned)(((ct) * 16 + crow) * 256 + cchunk * 16);       \
    asm volatile("global_load_async_to_lds_b128 %0, %1, %2"                    \
                 :: "v"(dst_), "v"(voff_), "s"(sbase) : "memory");             \
  } while (0)
#define WAIT_ASYNC() asm volatile("s_wait_asynccnt 0" ::: "memory")

  float dacc0 = 0.0f, spos0 = 0.0f;
  float dacc1 = 0.0f, spos1 = 0.0f;

  ISSUE_COPY(0, 0);
  WAIT_ASYNC();
  __syncthreads();

  for (int ct = 0; ct < NT; ++ct) {
    const int p = ct & 1;
    if (ct + 1 < NT) ISSUE_COPY(ct + 1, 1 - p);   // overlap copy with compute

    // A fragments (protos) from LDS, shared by both accumulators:
    // lane holds proto row ln; per step s the v16 elements are
    // K = 32s+8h+[0..8) then K = 32s+16+8h+[0..8).
    const unsigned short* arow = &ldsP[p][ln * LDS_ROW + 8 * h];
    v8f acc0 = {}, acc1 = {};
#pragma unroll
    for (int s = 0; s < 4; ++s) {
      v8u alo = *(const v8u*)(arow + 32 * s);
      v8u ahi = *(const v8u*)(arow + 32 * s + 16);
      v16u au;
#pragma unroll
      for (int e = 0; e < 8; ++e) { au[e] = alo[e]; au[8 + e] = ahi[e]; }
      v16bf a = __builtin_bit_cast(v16bf, au);
      acc0 = __builtin_amdgcn_wmma_f32_16x16x32_bf16(
          false, a, false, __builtin_bit_cast(v16bf, qfr0[s]),
          (short)0, acc0, false, false);
      acc1 = __builtin_amdgcn_wmma_f32_16x16x32_bf16(
          false, a, false, __builtin_bit_cast(v16bf, qfr1[s]),
          (short)0, acc1, false, false);
    }

    // ---- Per-lane epilogue: acc[r] = sim(query, proto r+8h of tile);
    // rows r+8h are the 8 prototypes of concept ct*2+h. No max-subtraction
    // needed: |gamma*s| <= ~20 stays comfortably in fp32 range.
    const int cid = ct * 2 + h;
    {
      float es = 0.0f, wsm = 0.0f;
#pragma unroll
      for (int r = 0; r < 8; ++r) {
        float sv = acc0[r];
        float e  = __builtin_amdgcn_exp2f(sv * kGL2E);   // exp(gamma*s)
        es  += e;
        wsm  = __builtin_fmaf(sv, e, wsm);
      }
      float simc = wsm * __builtin_amdgcn_rcpf(es);
      dacc0 += __builtin_amdgcn_exp2f(simc * kGL2E);     // exp(lambda*sim_c)
      if (cid == (q0 & (Kc - 1))) spos0 = simc;          // diagonal sim[b,k,k]
    }
    {
      float es = 0.0f, wsm = 0.0f;
#pragma unroll
      for (int r = 0; r < 8; ++r) {
        float sv = acc1[r];
        float e  = __builtin_amdgcn_exp2f(sv * kGL2E);
        es  += e;
        wsm  = __builtin_fmaf(sv, e, wsm);
      }
      float simc = wsm * __builtin_amdgcn_rcpf(es);
      dacc1 += __builtin_amdgcn_exp2f(simc * kGL2E);
      if (cid == (q1 & (Kc - 1))) spos1 = simc;
    }

    WAIT_ASYNC();       // next buffer's slice (this wave) has landed
    __syncthreads();    // publish copies; fence reads of buf p before overwrite
  }

  // ---- Finish: merge the two concept halves (lanes L, L^16) per query,
  // masked loss on lanes 0..15, wave sum, one atomic per wave.
  float d0  = dacc0 + swzx<16>(dacc0);
  float sp0 = spos0 + swzx<16>(spos0);
  float d1  = dacc1 + swzx<16>(dacc1);
  float sp1 = spos1 + swzx<16>(spos1);
  float lsum = 0.0f;
  if (lane < 16) {
    if (labels[q0] == 1)
      lsum += __logf(d0 + kEps) - kLambda * (sp0 + kMargin);
    if (labels[q1] == 1)
      lsum += __logf(d1 + kEps) - kLambda * (sp1 + kMargin);
  }
  lsum += dppx<1>(lsum);
  lsum += dppx<2>(lsum);
  lsum += dppx<4>(lsum);
  lsum += swzx<8>(lsum);
  if (lane == 0) atomicAdd(ws, lsum);

#undef ISSUE_COPY
#undef WAIT_ASYNC
}

__global__ void mpcl_init(float* __restrict__ ws) {
  if (threadIdx.x == 0) ws[0] = 0.0f;
}

__global__ __launch_bounds__(256) void mpcl_finalize(
    const int* __restrict__ labels, const float* __restrict__ ws,
    float* __restrict__ out) {
  __shared__ int red[256];
  int cnt = 0;
  for (int i = threadIdx.x; i < NQ; i += 256) cnt += (labels[i] == 1) ? 1 : 0;
  red[threadIdx.x] = cnt;
  __syncthreads();
  for (int off = 128; off > 0; off >>= 1) {
    if (threadIdx.x < off) red[threadIdx.x] += red[threadIdx.x + off];
    __syncthreads();
  }
  if (threadIdx.x == 0) {
    float np  = (float)red[0];
    float tot = ws[0];
    out[0] = (np > 0.0f) ? (tot / np) : tot;
  }
}

extern "C" void kernel_launch(void* const* d_in, const int* in_sizes, int n_in,
                              void* d_out, int out_size, void* d_ws, size_t ws_size,
                              hipStream_t stream) {
  const float* vproj  = (const float*)d_in[0];   // (B,K,D) fp32
  const int*   labels = (const int*)d_in[1];     // (B,K) int32 (jax default x64-off)
  const float* protos = (const float*)d_in[2];   // (K,M,D) fp32
  float* ws  = (float*)d_ws;
  float* out = (float*)d_out;

  mpcl_init<<<1, 32, 0, stream>>>(ws);
  mpcl_convert<<<(NP * Dd) / (256 * 8), 256, 0, stream>>>(protos);
  mpcl_main<<<NQ / 256, 256, 0, stream>>>(vproj, labels, ws);
  mpcl_finalize<<<1, 256, 0, stream>>>(labels, ws, out);
}
